// TropicalSSMLayer_62302795596525
// MI455X (gfx1250) — matrix operand order repaired and checked by reading, not compiled, not run on hardware
//
#include <hip/hip_runtime.h>
#include <hip/hip_bf16.h>
#include <math.h>

// ---------------------------------------------------------------------------
// TropicalSSMLayer fused implementation for gfx1250 (MI455X).
// Key identity: attention logits are key-only => causal softmax == online-
// softmax prefix scan (flash-style running max), O(T) instead of O(T^2).
// GEMMs run on v_wmma_f32_16x16x32_f16 (f16 in, f32 accumulate).
// Scan is a 3-level chunked associative scan: sequential depth 128, not 2048.
// ---------------------------------------------------------------------------

typedef _Float16 v16h __attribute__((ext_vector_type(16)));
typedef float    v8f  __attribute__((ext_vector_type(8)));

#define Bsz  4
#define Tseq 2048
#define Dmod 512
#define Hh   4
#define DKk  16
#define HDK  64
#define NTOK (Bsz * Tseq)   // 8192 tokens
#define NCAT 256            // q|k|v|gpre concatenated
#define NCH  16             // chunks per (b,h)
#define CS   128            // chunk size (Tseq / NCH)

union FragU { uint4 u[2]; v16h h; };

// ---------------------------------------------------------------------------
// Kernel 0: convert weights fp32 -> f16, pre-swizzled into WMMA B-fragment
// order.  B (KxN=32x16) per-lane layout used: lanes 0-15: N=lane, halves j ->
// K = ktBase + (j&7) + 16*(j>>3); lanes 16-31: N=lane-16, K = that + 8.
// Packed index: ((kt*NT + nt)*32 + lane)*16 + j  (contiguous 32B per lane).
// ---------------------------------------------------------------------------
__global__ void pack_weights(const float* __restrict__ Wq, const float* __restrict__ Wk,
                             const float* __restrict__ Wv, const float* __restrict__ Wg,
                             const float* __restrict__ Wo,
                             _Float16* __restrict__ wcat, _Float16* __restrict__ woh) {
    int gid = blockIdx.x * blockDim.x + threadIdx.x;
    if (gid < 512 * 256) {                       // Wq|Wk|Wv|Wg : K=512, N=256
        int j    = gid & 15;
        int lane = (gid >> 4) & 31;
        int nt   = (gid >> 9) & 15;
        int kt   = gid >> 13;                    // 0..15
        int N    = nt * 16 + (lane & 15);
        int K    = kt * 32 + (lane >> 4) * 8 + (j & 7) + (j >> 3) * 16;
        const float* src = (N < 64) ? Wq : (N < 128) ? Wk : (N < 192) ? Wv : Wg;
        wcat[gid] = (_Float16)src[K * 64 + (N & 63)];
    } else {                                     // Wo : K=64, N=512
        int idx = gid - 512 * 256;
        if (idx < 64 * 512) {
            int j    = idx & 15;
            int lane = (idx >> 4) & 31;
            int nt   = (idx >> 9) & 31;
            int kt   = idx >> 14;                // 0..1
            int N    = nt * 16 + (lane & 15);
            int K    = kt * 32 + (lane >> 4) * 8 + (j & 7) + (j >> 3) * 16;
            woh[idx] = (_Float16)Wo[K * 512 + N];
        }
    }
}

// ---------------------------------------------------------------------------
// Kernel 1: projections P[token][0:256] = x @ [Wq|Wk|Wv|Wg]  (fp32 out).
// One block = 16 tokens (M-tile), 8 waves; wave w owns N-tiles {w, w+8}.
// x tile staged in LDS as f16 once; A-fragment layout per ISA 16-bit A table.
// ---------------------------------------------------------------------------
__global__ void proj_gemm(const float* __restrict__ x, const _Float16* __restrict__ wcat,
                          float* __restrict__ P) {
    __shared__ _Float16 lA[16 * 512];            // 16 KB
    const int mBase = blockIdx.x * 16;
    const int tid   = threadIdx.x;

    // stage + convert x tile (16x512 fp32 -> f16)
    {
        int row = tid >> 4;
        int col = (tid & 15) * 32;
        const float* xr = x + (size_t)(mBase + row) * Dmod + col;
        _Float16* dst   = lA + row * 512 + col;
#pragma unroll
        for (int j = 0; j < 32; j += 4) {
            float4 f = *(const float4*)(xr + j);
            dst[j + 0] = (_Float16)f.x; dst[j + 1] = (_Float16)f.y;
            dst[j + 2] = (_Float16)f.z; dst[j + 3] = (_Float16)f.w;
        }
    }
    __syncthreads();

    const int wave = tid >> 5;
    const int lane = tid & 31;
    const int M    = lane & 15;
    const int hi   = lane >> 4;
    const int nt0  = wave, nt1 = wave + 8;

    v8f acc0 = {}; v8f acc1 = {};
    for (int kt = 0; kt < 16; ++kt) {
        FragU a;
        int kb = kt * 32 + hi * 8;               // A: halves 0-7 -> K=kb..kb+7
        a.u[0] = *(const uint4*)(lA + M * 512 + kb);
        a.u[1] = *(const uint4*)(lA + M * 512 + kb + 16); // halves 8-15 -> K=kb+16..
        FragU b0, b1;
        const _Float16* bp0 = wcat + ((size_t)(kt * 16 + nt0) * 32 + lane) * 16;
        const _Float16* bp1 = wcat + ((size_t)(kt * 16 + nt1) * 32 + lane) * 16;
        b0.u[0] = *(const uint4*)bp0;       b0.u[1] = *(const uint4*)(bp0 + 8);
        b1.u[0] = *(const uint4*)bp1;       b1.u[1] = *(const uint4*)(bp1 + 8);
        acc0 = __builtin_amdgcn_wmma_f32_16x16x32_f16(false, a.h, false, b0.h,
                                                      (short)0, acc0, false, false);
        acc1 = __builtin_amdgcn_wmma_f32_16x16x32_f16(false, a.h, false, b1.h,
                                                      (short)0, acc1, false, false);
    }
    // C/D layout: VGPR r -> row M = r + 8*hi, col N = lane&15
#pragma unroll
    for (int r = 0; r < 8; ++r) {
        int row = mBase + r + 8 * hi;
        P[(size_t)row * NCAT + nt0 * 16 + M] = acc0[r];
        P[(size_t)row * NCAT + nt1 * 16 + M] = acc1[r];
    }
}

// ---------------------------------------------------------------------------
// Kernel 2: bid = <q,k>/sqrt(16);  a = -softplus(x @ Wa + ba).
// One thread per (token, head).  Stored as (B,H,T) for the scan.
// ---------------------------------------------------------------------------
__global__ void bid_a_kernel(const float* __restrict__ x, const float* __restrict__ P,
                             const float* __restrict__ Wa, const float* __restrict__ ba,
                             float* __restrict__ bidw, float* __restrict__ aw) {
    int g     = blockIdx.x * blockDim.x + threadIdx.x;  // < NTOK*H
    int token = g >> 2;
    int h     = g & 3;
    const float* prow = P + (size_t)token * NCAT;
    float s = 0.f;
#pragma unroll
    for (int d = 0; d < DKk; ++d)
        s += prow[h * 16 + d] * prow[64 + h * 16 + d];
    float bid = s * 0.25f;                               // 1/sqrt(16)

    float ap = ba[h];
    const float* xr = x + (size_t)token * Dmod;
    for (int k = 0; k < Dmod; ++k)
        ap += xr[k] * Wa[k * Hh + h];
    float a = -((ap > 20.f) ? ap : log1pf(__expf(ap)));  // -softplus

    int b = token / Tseq, t = token % Tseq;
    size_t o = ((size_t)b * Hh + h) * Tseq + t;
    bidw[o] = bid;
    aw[o]   = a;
}

// ---------------------------------------------------------------------------
// Kernel 3: chunked associative online-softmax scan.  One block per (b,h);
// 16 waves, wave c owns chunk t in [c*128, c*128+128).  The combine op on
// (m, Z, N[16]) is associative, so:
//   A) lane-scan for within-chunk cumsum(a), lg -> LDS
//   B) exclusive scan of chunk sumA (depth 16)
//   C) per-chunk summaries m_c, Z_c (shfl_xor reductions), N_c (lane = d x
//      t-half remap, depth 64, loads independent of the accumulate chain)
//   D) exclusive cross-chunk combine (depth 16)
//   E) within-chunk inclusive pass seeded by D (depth 128) + sigmoid gate
// Sequential depth ~128+16 instead of 2048; 256 active waves instead of 16.
// ---------------------------------------------------------------------------
__global__ void scan_kernel(const float* __restrict__ P, const float* __restrict__ bidw,
                            const float* __restrict__ aw, const float* __restrict__ bg,
                            float* __restrict__ yg) {
    __shared__ float lgS[Tseq];        // adjusted logits (8 KB)
    __shared__ float sAc[NCH], cbS[NCH];
    __shared__ float mC[NCH], zC[NCH], nC[NCH][DKk];
    __shared__ float MpS[NCH], ZpS[NCH], NpS[NCH][DKk];

    const int bh   = blockIdx.x;       // b*H + h
    const int b    = bh >> 2;
    const int h    = bh & 3;
    const int tid  = threadIdx.x;
    const int c    = tid >> 5;         // wave = chunk id (0..15)
    const int lane = tid & 31;
    const int cbase = c * CS;
    const float* bidr = bidw + (size_t)bh * Tseq;
    const float* ar   = aw   + (size_t)bh * Tseq;

    // ---- A: within-chunk cumsum(a) via lane scan; lg (minus chunk base) ----
    float4 av = *(const float4*)(ar   + cbase + lane * 4);
    float4 bv = *(const float4*)(bidr + cbase + lane * 4);
    float c0 = av.x, c1 = c0 + av.y, c2 = c1 + av.z, c3 = c2 + av.w;
    float s = c3;                      // inclusive Hillis-Steele over lanes
#pragma unroll
    for (int off = 1; off < 32; off <<= 1) {
        float t2 = __shfl_up(s, off, 32);
        if (lane >= off) s += t2;
    }
    float excl = s - c3;
    int t0 = cbase + lane * 4;
    lgS[t0 + 0] = bv.x - (excl + c0);
    lgS[t0 + 1] = bv.y - (excl + c1);
    lgS[t0 + 2] = bv.z - (excl + c2);
    lgS[t0 + 3] = bv.w - (excl + c3);
    if (lane == 31) sAc[c] = s;        // chunk total sum of a
    __syncthreads();

    // ---- B: exclusive scan of chunk sums -> cumA base per chunk ----
    if (tid == 0) {
        float run = 0.f;
        for (int i = 0; i < NCH; ++i) { cbS[i] = run; run += sAc[i]; }
    }
    __syncthreads();

    // ---- C: adjust lg, per-chunk (m, Z) reductions, N via d x half remap ----
    float base = cbS[c];
    float l0 = lgS[t0 + 0] - base, l1 = lgS[t0 + 1] - base;
    float l2 = lgS[t0 + 2] - base, l3 = lgS[t0 + 3] - base;
    lgS[t0 + 0] = l0; lgS[t0 + 1] = l1; lgS[t0 + 2] = l2; lgS[t0 + 3] = l3;
    float mloc = fmaxf(fmaxf(l0, l1), fmaxf(l2, l3));
#pragma unroll
    for (int off = 16; off >= 1; off >>= 1)
        mloc = fmaxf(mloc, __shfl_xor(mloc, off, 32));
    float zloc = __expf(l0 - mloc) + __expf(l1 - mloc) +
                 __expf(l2 - mloc) + __expf(l3 - mloc);
#pragma unroll
    for (int off = 16; off >= 1; off >>= 1)
        zloc += __shfl_xor(zloc, off, 32);
    if (lane == 0) { mC[c] = mloc; zC[c] = zloc; }

    {   // N_c[d] = sum_t e^{lg-m_c} v[t][d]; lane = d + 16*half, 64 t each
        int d = lane & 15, g2 = lane >> 4;
        float np = 0.f;
        const float* vp = P + 128 + h * 16 + d;
        for (int i = 0; i < 64; ++i) {
            int t = cbase + g2 * 64 + i;
            float e = __expf(lgS[t] - mloc);
            np += e * vp[(size_t)(b * Tseq + t) * NCAT];
        }
        np += __shfl_xor(np, 16, 32);
        if (g2 == 0) nC[c][d] = np;
    }
    __syncthreads();

    // ---- D: exclusive cross-chunk combine of (m, Z, N) ----
    if (tid < 16) {
        int d = tid;
        float M = -INFINITY, Z = 0.f, Nd = 0.f;
        for (int i = 0; i < NCH; ++i) {
            MpS[i] = M; ZpS[i] = Z; NpS[i][d] = Nd;   // same M,Z from all d: benign
            float mn  = fmaxf(M, mC[i]);
            float sc1 = __expf(M - mn), sc2 = __expf(mC[i] - mn);
            Z  = Z  * sc1 + zC[i]    * sc2;
            Nd = Nd * sc1 + nC[i][d] * sc2;
            M  = mn;
        }
    }
    __syncthreads();

    // ---- E: within-chunk inclusive pass seeded with exclusive prefix ----
    {
        int d = lane & 15, g2 = lane >> 4;            // halves redundant
        float bgv = bg[h * 16 + d];
        float M = MpS[c], Z = ZpS[c], Nd = NpS[c][d];
        const float* vp = P + 128 + h * 16 + d;
        const float* gp = P + 192 + h * 16 + d;
        for (int i = 0; i < CS; ++i) {
            int t = cbase + i;
            size_t tok = (size_t)(b * Tseq + t);
            float lg = lgS[t];
            float mn = fmaxf(M, lg);
            float sc = __expf(M - mn);
            float e  = __expf(lg - mn);
            Z  = Z * sc + e;
            Nd = Nd * sc + e * vp[tok * NCAT];
            M  = mn;
            if (g2 == 0) {
                float gpre = gp[tok * NCAT] + bgv;
                float gate = 1.f / (1.f + __expf(-gpre));
                yg[tok * HDK + h * 16 + d] = (Nd / Z) * gate;
            }
        }
    }
}

// ---------------------------------------------------------------------------
// Kernel 4: out = yg (B*T x 64) @ Wo (64 x 512) via WMMA.  One block = 16
// tokens; wave w owns 4 N-tiles; K = 64 -> two wmma steps per tile.
// ---------------------------------------------------------------------------
__global__ void out_gemm(const float* __restrict__ yg, const _Float16* __restrict__ woh,
                         float* __restrict__ out) {
    __shared__ _Float16 lA[16 * 64];               // 2 KB
    const int mBase = blockIdx.x * 16;
    const int tid   = threadIdx.x;
    {
        int idx = tid * 4;                         // 256 threads * 4 = 1024 elems
        int row = idx >> 6;
        int col = idx & 63;
        float4 f = *(const float4*)(yg + (size_t)(mBase + row) * HDK + col);
        _Float16* d2 = lA + row * 64 + col;
        d2[0] = (_Float16)f.x; d2[1] = (_Float16)f.y;
        d2[2] = (_Float16)f.z; d2[3] = (_Float16)f.w;
    }
    __syncthreads();

    const int wave = tid >> 5;
    const int lane = tid & 31;
    const int M    = lane & 15;
    const int hi   = lane >> 4;

    v8f acc[4] = {};
    for (int kt = 0; kt < 2; ++kt) {
        FragU a;
        int kb = kt * 32 + hi * 8;
        a.u[0] = *(const uint4*)(lA + M * 64 + kb);
        a.u[1] = *(const uint4*)(lA + M * 64 + kb + 16);
#pragma unroll
        for (int i = 0; i < 4; ++i) {
            int nt = wave * 4 + i;
            FragU bf;
            const _Float16* bp = woh + ((size_t)(kt * 32 + nt) * 32 + lane) * 16;
            bf.u[0] = *(const uint4*)bp;
            bf.u[1] = *(const uint4*)(bp + 8);
            acc[i] = __builtin_amdgcn_wmma_f32_16x16x32_f16(false, a.h, false, bf.h,
                                                            (short)0, acc[i], false, false);
        }
    }
#pragma unroll
    for (int i = 0; i < 4; ++i) {
        int nb = (wave * 4 + i) * 16;
#pragma unroll
        for (int r = 0; r < 8; ++r)
            out[(size_t)(mBase + r + 8 * hi) * Dmod + nb + M] = acc[i][r];
    }
}

// ---------------------------------------------------------------------------
extern "C" void kernel_launch(void* const* d_in, const int* in_sizes, int n_in,
                              void* d_out, int out_size, void* d_ws, size_t ws_size,
                              hipStream_t stream) {
    const float* x  = (const float*)d_in[0];
    const float* Wq = (const float*)d_in[1];
    const float* Wk = (const float*)d_in[2];
    const float* Wv = (const float*)d_in[3];
    const float* Wa = (const float*)d_in[4];
    const float* ba = (const float*)d_in[5];
    const float* Wg = (const float*)d_in[6];
    const float* bg = (const float*)d_in[7];
    const float* Wo = (const float*)d_in[8];
    float* out = (float*)d_out;

    // workspace layout (fp32 then packed f16; all 16B aligned)
    float* P    = (float*)d_ws;                          // NTOK*256   = 8 MB
    float* bidw = P    + (size_t)NTOK * NCAT;            // B*H*T      = 128 KB
    float* aw   = bidw + (size_t)Bsz * Hh * Tseq;        // B*H*T      = 128 KB
    float* yg   = aw   + (size_t)Bsz * Hh * Tseq;        // NTOK*64    = 2 MB
    _Float16* wcat = (_Float16*)(yg + (size_t)NTOK * HDK);   // 512*256 halves
    _Float16* woh  = wcat + (size_t)512 * 256;               // 64*512 halves

    pack_weights<<<640, 256, 0, stream>>>(Wq, Wk, Wv, Wg, Wo, wcat, woh);
    proj_gemm   <<<NTOK / 16, 256, 0, stream>>>(x, wcat, P);
    bid_a_kernel<<<(NTOK * Hh) / 256, 256, 0, stream>>>(x, P, Wa, ba, bidw, aw);
    scan_kernel <<<Bsz * Hh, NCH * 32, 0, stream>>>(P, bidw, aw, bg, yg);
    out_gemm    <<<NTOK / 16, 256, 0, stream>>>(yg, woh, out);
}
